// DNC_6442450944383
// MI455X (gfx1250) — matrix operand order repaired and checked by reading, not compiled, not run on hardware
//
#include <hip/hip_runtime.h>
#include <math.h>

// ---------------- types for CDNA5 WMMA ----------------
typedef __attribute__((ext_vector_type(16))) __bf16 bf16x16;
typedef __attribute__((ext_vector_type(8)))  __bf16 bf16x8;
typedef __attribute__((ext_vector_type(8)))  float  f32x8;

// ---------------- problem constants ----------------
constexpr int B_   = 32;
constexpr int L_   = 256;
constexpr int E_   = 256;
constexpr int H_   = 512;
constexpr int N_   = 128;
constexpr int W_   = 64;
constexpr int OUT_ = 256;
constexpr int IFP_ = 208;   // IFACE=198 padded to 13 tiles of 16
constexpr int NWG  = 32;
constexpr int KA   = 832;   // x(256) | rv(64) | h(512)

// ---------------- workspace layout (bytes) ----------------
constexpr size_t OFF_BAR   = 0;
constexpr size_t OFF_MEM   = 256;
constexpr size_t OFF_LINK  = OFF_MEM   + (size_t)B_*N_*W_*4;   // 1 MB
constexpr size_t OFF_USAGE = OFF_LINK  + (size_t)B_*N_*N_*4;   // 2 MB
constexpr size_t OFF_PREC  = OFF_USAGE + (size_t)B_*N_*4;
constexpr size_t OFF_RW    = OFF_PREC  + (size_t)B_*N_*4;
constexpr size_t OFF_HBF   = OFF_RW    + (size_t)B_*N_*4;
constexpr size_t OFF_RVBF  = OFF_HBF   + (size_t)B_*H_*2;
constexpr size_t OFF_IFS   = OFF_RVBF  + (size_t)B_*W_*2;
constexpr size_t ZERO_END  = OFF_IFS   + (size_t)B_*IFP_*4;
constexpr size_t OFF_WIBF  = (ZERO_END + 255) & ~(size_t)255;
constexpr size_t OFF_WOBF  = OFF_WIBF  + (size_t)IFP_*H_*2;
constexpr size_t WS_NEED   = OFF_WOBF  + (size_t)OUT_*(H_+W_)*2;

// ---------------- LDS layout (bytes) ----------------
constexpr int A_STRIDE = 848;                                   // K stride (elems), 16B aligned rows
constexpr size_t LDS_W_OFF    = 0;                              // 64 cols x 848 bf16 = 108544
constexpr size_t LDS_A_OFF    = (size_t)64 * A_STRIDE * 2;
constexpr size_t LDS_C_OFF    = LDS_A_OFF + (size_t)32 * A_STRIDE * 2;  // c state + gate bias
constexpr size_t LDS_MISC_OFF = LDS_C_OFF + 2560;
constexpr size_t LDS_TOTAL    = LDS_MISC_OFF + (size_t)19104 * 4;       // ~242 KB < 320 KB

// ---------------- helpers ----------------
__device__ __forceinline__ float sigmoidf_(float x) { return 1.f / (1.f + expf(-x)); }

__device__ __forceinline__ bf16x16 frag_a_lds(const __bf16* A, int m0, int k0, int lane) {
  // 16-bit A 16x32: lanes0-15 row=lane, K=k0..k0+15; lanes16-31 K=k0+16..k0+31
  const __bf16* p = A + (m0 + (lane & 15)) * A_STRIDE + k0 + ((lane >> 4) << 4);
  bf16x8 lo = *(const bf16x8*)p;
  bf16x8 hi = *(const bf16x8*)(p + 8);
  bf16x16 r;
#pragma unroll
  for (int i = 0; i < 8; ++i) { r[i] = lo[i]; r[i + 8] = hi[i]; }
  return r;
}
__device__ __forceinline__ bf16x16 frag_b_lds(const __bf16* Bm, int n0, int k0, int lane) {
  // 16-bit B 32x16: lanes0-15 col=lane K=k0..k0+15; lanes16-31 K=k0+16..k0+31
  const __bf16* p = Bm + (n0 + (lane & 15)) * A_STRIDE + k0 + ((lane >> 4) << 4);
  bf16x8 lo = *(const bf16x8*)p;
  bf16x8 hi = *(const bf16x8*)(p + 8);
  bf16x16 r;
#pragma unroll
  for (int i = 0; i < 8; ++i) { r[i] = lo[i]; r[i + 8] = hi[i]; }
  return r;
}
__device__ __forceinline__ bf16x16 frag_b_glb(const __bf16* Bm, int n0, int k0, int lane, int stride) {
  const __bf16* p = Bm + (size_t)(n0 + (lane & 15)) * stride + k0 + ((lane >> 4) << 4);
  bf16x8 lo = *(const bf16x8*)p;
  bf16x8 hi = *(const bf16x8*)(p + 8);
  bf16x16 r;
#pragma unroll
  for (int i = 0; i < 8; ++i) { r[i] = lo[i]; r[i + 8] = hi[i]; }
  return r;
}

__device__ __forceinline__ float blk_sum(float v, float* buf) {
  int tid = threadIdx.x;
  buf[tid] = v; __syncthreads();
#pragma unroll
  for (int s = 128; s > 0; s >>= 1) { if (tid < s) buf[tid] += buf[tid + s]; __syncthreads(); }
  float r = buf[0]; __syncthreads();
  return r;
}
__device__ __forceinline__ float blk_max(float v, float* buf) {
  int tid = threadIdx.x;
  buf[tid] = v; __syncthreads();
#pragma unroll
  for (int s = 128; s > 0; s >>= 1) { if (tid < s) buf[tid] = fmaxf(buf[tid], buf[tid + s]); __syncthreads(); }
  float r = buf[0]; __syncthreads();
  return r;
}

__device__ __forceinline__ void grid_barrier(unsigned* arrive, unsigned* release, unsigned gen) {
  __syncthreads();
  if (threadIdx.x == 0) {
    __threadfence();
    unsigned old = atomicAdd(arrive, 1u);
    if (old == (unsigned)(NWG - 1)) {
      atomicExch(arrive, 0u);
      atomicAdd(release, 1u);
    } else {
      while (atomicAdd(release, 0u) < gen) { __builtin_amdgcn_s_sleep(4); }
    }
    __threadfence();
  }
  __syncthreads();
}

// ---------------- prep: pack Wi / Wo to bf16, K-permuted to panel space ----------------
__global__ void dnc_prep(const float* __restrict__ Wi, const float* __restrict__ Wo,
                         __bf16* __restrict__ Wi_bf, __bf16* __restrict__ Wo_bf) {
  int i = blockIdx.x * blockDim.x + threadIdx.x;
  int stride = gridDim.x * blockDim.x;
  for (int idx = i; idx < IFP_ * H_; idx += stride) {
    int n = idx >> 9, k = idx & 511;
    Wi_bf[idx] = (__bf16)((n < 198) ? Wi[n * H_ + k] : 0.f);
  }
  for (int idx = i; idx < OUT_ * (H_ + W_); idx += stride) {
    int n = idx / 576, k = idx - n * 576;
    // panel K: 0..63 = read_vec (Wo cols 512..575), 64..575 = h (Wo cols 0..511)
    float v = (k < 64) ? Wo[n * 576 + 512 + k] : Wo[n * 576 + (k - 64)];
    Wo_bf[idx] = (__bf16)v;
  }
}

// ---------------- persistent DNC kernel ----------------
__global__ void __launch_bounds__(256, 1)
dnc_persistent(const float* __restrict__ x_emb, const float* __restrict__ W_ih,
               const float* __restrict__ W_hh, const float* __restrict__ b_ih,
               const float* __restrict__ b_hh, const float* __restrict__ bi,
               const float* __restrict__ bo, float* __restrict__ out,
               char* __restrict__ ws) {
  extern __shared__ __align__(16) char smem[];
  __bf16* Wbuf = (__bf16*)(smem + LDS_W_OFF);     // [4 tiles][16 cols][848]
  __bf16* Abuf = (__bf16*)(smem + LDS_A_OFF);     // [32 rows][848]  x|rv|h
  float*  cbuf = (float*)(smem + LDS_C_OFF);      // [512] c-state slice
  float*  gbias = cbuf + 512;                     // [64]
  float*  MF = (float*)(smem + LDS_MISC_OFF);
  float* gb       = MF;                 // [4][32][16] gate tiles  (aliases linkbuf)
  float* linkbuf  = MF;                 // [128][128]
  float* reda     = MF + 16384;
  float* redb     = MF + 16640;
  float* v_if     = MF + 16896;
  float* v_erase  = MF + 17104;
  float* v_wvec   = MF + 17168;
  float* v_key    = MF + 17232;
  float* v_us     = MF + 17296;
  float* v_sorted = MF + 17424;
  int*   v_sidx   = (int*)(MF + 17552);
  float* v_scan   = MF + 17680;
  float* v_allocw = MF + 17808;
  float* v_ww     = MF + 17936;
  float* v_prec   = MF + 18064;
  float* v_rwold  = MF + 18192;
  float* v_dot    = MF + 18320;
  float* v_norm   = MF + 18448;
  float* v_cw     = MF + 18576;
  float* v_bw     = MF + 18704;
  float* v_fw     = MF + 18832;
  float* v_rwn    = MF + 18960;
  float* sc       = MF + 19088;

  unsigned* arrive  = (unsigned*)(ws + OFF_BAR);
  unsigned* release = arrive + 1;
  float*  mem_g   = (float*)(ws + OFF_MEM);
  float*  link_g  = (float*)(ws + OFF_LINK);
  float*  usage_g = (float*)(ws + OFF_USAGE);
  float*  prec_g  = (float*)(ws + OFF_PREC);
  float*  rw_g    = (float*)(ws + OFF_RW);
  __bf16* h_bf_g  = (__bf16*)(ws + OFF_HBF);
  __bf16* rv_bf_g = (__bf16*)(ws + OFF_RVBF);
  float*  iface_g = (float*)(ws + OFF_IFS);
  const __bf16* Wi_bf = (const __bf16*)(ws + OFF_WIBF);
  const __bf16* Wo_bf = (const __bf16*)(ws + OFF_WOBF);

  const int tid  = threadIdx.x;
  const int lane = tid & 31;
  const int wv   = tid >> 5;
  const int wg   = blockIdx.x;

  // ---------- init: zero panel, load x_0, c=0, bias, weight slice -> LDS bf16 ----------
  for (int i = tid; i < 32 * A_STRIDE / 4; i += 256) ((unsigned long long*)Abuf)[i] = 0ull;
  for (int i = tid; i < B_ * E_; i += 256) {
    int b = i >> 8, e = i & 255;
    Abuf[b * A_STRIDE + e] = (__bf16)x_emb[b * (L_ * E_) + e];
  }
  for (int i = tid; i < 512; i += 256) cbuf[i] = 0.f;
  if (tid < 64) {
    int q = tid >> 4, cc = tid & 15;
    int row = q * H_ + wg * 16 + cc;
    gbias[tid] = b_ih[row] + b_hh[row];
  }
  for (int i = tid; i < 64 * KA; i += 256) {
    int rr = i / KA, k = i - rr * KA;
    int q = rr >> 4, cc = rr & 15;
    int row = q * H_ + wg * 16 + cc;
    float v = (k < 320) ? W_ih[row * 320 + k] : W_hh[row * H_ + (k - 320)];
    Wbuf[rr * A_STRIDE + k] = (__bf16)v;
  }
  __syncthreads();

  unsigned gen = 0;
  for (int t = 0; t < L_; ++t) {
    // ================= P2: gates GEMM (bf16 WMMA) + LSTM =================
    {
      const int mt = wv & 1, q = wv >> 1;
      f32x8 acc = {};
      for (int kt = 0; kt < 26; ++kt) {
        bf16x16 a  = frag_a_lds(Abuf, mt * 16, kt * 32, lane);
        bf16x16 bb = frag_b_lds(Wbuf, q * 16, kt * 32, lane);
        acc = __builtin_amdgcn_wmma_f32_16x16x32_bf16(false, a, false, bb, (short)0, acc, false, false);
      }
      const int col = lane & 15, brow = (lane >> 4) << 3;
#pragma unroll
      for (int r = 0; r < 8; ++r) gb[q * 512 + (mt * 16 + brow + r) * 16 + col] = acc[r];
    }
    __syncthreads();
    for (int idx = tid; idx < 512; idx += 256) {
      int b = idx >> 4, j = idx & 15;
      float ig = gb[idx]            + gbias[j];
      float fg = gb[512 + idx]      + gbias[16 + j];
      float gg = gb[1024 + idx]     + gbias[32 + j];
      float og = gb[1536 + idx]     + gbias[48 + j];
      float c  = cbuf[idx];
      float cn = sigmoidf_(fg) * c + sigmoidf_(ig) * tanhf(gg);
      cbuf[idx] = cn;
      float h = sigmoidf_(og) * tanhf(cn);
      h_bf_g[b * H_ + wg * 16 + j] = (__bf16)h;
    }
    grid_barrier(arrive, release, ++gen);

    // ================= P3: refresh panel h-cols; iface GEMM =================
    for (int i = tid; i < 32 * 128; i += 256) {
      int b = i >> 7, jj = i & 127;
      ((unsigned long long*)(Abuf + b * A_STRIDE + 320))[jj] =
          ((const unsigned long long*)(h_bf_g + b * H_))[jj];
    }
    __syncthreads();
    if (wg < 13 && wv < 2) {
      const int mt = wv;
      f32x8 acc = {};
      for (int kt = 0; kt < 16; ++kt) {
        bf16x16 a  = frag_a_lds(Abuf, mt * 16, 320 + kt * 32, lane);
        bf16x16 bb = frag_b_glb(Wi_bf, wg * 16, kt * 32, lane, H_);
        acc = __builtin_amdgcn_wmma_f32_16x16x32_bf16(false, a, false, bb, (short)0, acc, false, false);
      }
      int colg = wg * 16 + (lane & 15);
      float bv = (colg < 198) ? bi[colg] : 0.f;
      const int brow = (lane >> 4) << 3;
#pragma unroll
      for (int r = 0; r < 8; ++r) iface_g[(mt * 16 + brow + r) * IFP_ + colg] = acc[r] + bv;
    }
    grid_barrier(arrive, release, ++gen);

    // ================= P4: per-batch DNC memory machinery (WG = batch) =================
    {
      const int b_ = wg;
      if (tid < IFP_) v_if[tid] = iface_g[b_ * IFP_ + tid];
      if (tid < N_) {
        v_us[tid]    = usage_g[b_ * N_ + tid];
        v_prec[tid]  = prec_g[b_ * N_ + tid];
        v_rwold[tid] = rw_g[b_ * N_ + tid];
      }
      __syncthreads();
      if (tid < W_) {
        v_erase[tid] = sigmoidf_(v_if[tid]);
        v_wvec[tid]  = v_if[64 + tid];
        v_key[tid]   = v_if[130 + tid];
      }
      if (tid == 0) {
        sc[0] = sigmoidf_(v_if[128]) * sigmoidf_(v_if[129]);   // write_gate*alloc_gate
        sc[1] = log1pf(expf(v_if[194]));                       // read_strength
        float a0 = v_if[195], a1 = v_if[196], a2 = v_if[197];
        float mx = fmaxf(a0, fmaxf(a1, a2));
        float e0 = expf(a0 - mx), e1 = expf(a1 - mx), e2 = expf(a2 - mx);
        float s = e0 + e1 + e2;
        sc[2] = e0 / s; sc[3] = e1 / s; sc[4] = e2 / s;
      }
      __syncthreads();
      // stable argsort of usage via rank
      if (tid < N_) {
        float u = v_us[tid]; int r = 0;
        for (int k = 0; k < N_; ++k) { float uk = v_us[k]; r += (uk < u) || (uk == u && k < tid); }
        v_sidx[r] = tid; v_sorted[r] = u;
      }
      __syncthreads();
      if (tid < N_) v_scan[tid] = v_sorted[tid];
      __syncthreads();
      for (int s = 1; s < N_; s <<= 1) {      // Hillis-Steele inclusive product scan
        float tv = 1.f;
        if (tid < N_ && tid >= s) tv = v_scan[tid - s];
        __syncthreads();
        if (tid < N_ && tid >= s) v_scan[tid] *= tv;
        __syncthreads();
      }
      if (tid < N_) {
        float cpe = (tid == 0) ? 1.f : v_scan[tid - 1];
        v_allocw[v_sidx[tid]] = (1.f - v_sorted[tid]) * cpe;
      }
      __syncthreads();
      if (tid < N_) v_ww[tid] = sc[0] * v_allocw[tid];
      __syncthreads();
      float sww = blk_sum((tid < N_) ? v_ww[tid] : 0.f, reda);
      float kn2 = blk_sum((tid < W_) ? v_key[tid] * v_key[tid] : 0.f, reda);
      float keynorm = sqrtf(kn2) + 1e-8f;
      // mem update + dot/norm
      {
        const int n = tid >> 1, wh = (tid & 1) << 5;
        const float wwn = v_ww[n];
        float* Mrow = mem_g + ((size_t)(b_ * N_ + n) << 6);
        float pd = 0.f, pn = 0.f;
        for (int w = wh; w < wh + 32; ++w) {
          float m = Mrow[w];
          m = m * (1.f - wwn * v_erase[w]) + wwn * v_wvec[w];
          Mrow[w] = m;
          pd += v_key[w] * m; pn += m * m;
        }
        reda[tid] = pd; redb[tid] = pn; __syncthreads();
        if ((tid & 1) == 0) { v_dot[n] = reda[tid] + reda[tid + 1]; v_norm[n] = redb[tid] + redb[tid + 1]; }
        __syncthreads();
      }
      if (tid < N_) { float u = v_us[tid]; usage_g[b_ * N_ + tid] = u + (1.f - u) * v_ww[tid]; }
      // content weights (cosine softmax)
      float lg = (tid < N_) ? (v_dot[tid] / (keynorm * (sqrtf(v_norm[tid]) + 1e-8f)) * sc[1]) : -3.0e38f;
      float mx = blk_max(lg, reda);
      float ex = (tid < N_) ? expf(lg - mx) : 0.f;
      float es = blk_sum(ex, reda);
      if (tid < N_) v_cw[tid] = ex / es;
      __syncthreads();
      // link update + fw, then bw
      {
        const int n = tid >> 1, mh = (tid & 1) << 6;
        const float wwn = v_ww[n], pcn = v_prec[n];
        float* Lrow = link_g + ((size_t)(b_ * N_ + n) << 7);
        float pfw = 0.f;
        for (int m = mh; m < mh + 64; ++m) {
          float Lv = Lrow[m];
          float Ln = (1.f - wwn - v_ww[m]) * Lv + pcn * v_ww[m];
          if (m == n) Ln = 0.f;
          Lrow[m] = Ln;
          linkbuf[(n << 7) + m] = Ln;
          pfw += Ln * v_rwold[m];
        }
        reda[tid] = pfw; __syncthreads();
        if ((tid & 1) == 0) v_fw[n] = reda[tid] + reda[tid + 1];
        __syncthreads();
        const int m3 = tid >> 1, nh = (tid & 1) << 6;
        float pbw = 0.f;
        for (int n2 = nh; n2 < nh + 64; ++n2) pbw += linkbuf[(n2 << 7) + m3] * v_rwold[n2];
        reda[tid] = pbw; __syncthreads();
        if ((tid & 1) == 0) v_bw[m3] = reda[tid] + reda[tid + 1];
        __syncthreads();
      }
      float rwv = (tid < N_) ? (sc[2] * v_bw[tid] + sc[3] * v_fw[tid] + sc[4] * v_cw[tid] + 1e-8f) : 0.f;
      float rs = blk_sum(rwv, reda);
      if (tid < N_) { float rn = rwv / rs; v_rwn[tid] = rn; rw_g[b_ * N_ + tid] = rn; }
      if (tid < N_) prec_g[b_ * N_ + tid] = (1.f - sww) * v_prec[tid] + v_ww[tid];
      __syncthreads();
      // read vector
      {
        const int w = tid >> 2, nh = (tid & 3) << 5;
        const float* Mb = mem_g + ((size_t)b_ << 13);
        float p = 0.f;
        for (int n2 = nh; n2 < nh + 32; ++n2) p += v_rwn[n2] * Mb[(n2 << 6) + w];
        reda[tid] = p; __syncthreads();
        if ((tid & 3) == 0)
          rv_bf_g[(b_ << 6) + w] = (__bf16)(reda[tid] + reda[tid + 1] + reda[tid + 2] + reda[tid + 3]);
        __syncthreads();
      }
    }
    grid_barrier(arrive, release, ++gen);

    // ================= P5: refresh rv-cols; out GEMM; stage x_{t+1} =================
    for (int i = tid; i < 32 * 16; i += 256) {
      int b = i >> 4, jj = i & 15;
      ((unsigned long long*)(Abuf + b * A_STRIDE + 256))[jj] =
          ((const unsigned long long*)(rv_bf_g + b * W_))[jj];
    }
    __syncthreads();
    if (wg < 16 && wv < 2) {
      const int mt = wv;
      f32x8 acc = {};
      for (int kt = 0; kt < 18; ++kt) {
        bf16x16 a  = frag_a_lds(Abuf, mt * 16, 256 + kt * 32, lane);
        bf16x16 bb = frag_b_glb(Wo_bf, wg * 16, kt * 32, lane, H_ + W_);
        acc = __builtin_amdgcn_wmma_f32_16x16x32_bf16(false, a, false, bb, (short)0, acc, false, false);
      }
      int colg = wg * 16 + (lane & 15);
      float bv = bo[colg];
      const int brow = (lane >> 4) << 3;
#pragma unroll
      for (int r = 0; r < 8; ++r) {
        int b = mt * 16 + brow + r;
        out[(size_t)b * (L_ * OUT_) + t * OUT_ + colg] = acc[r] + bv;
      }
    }
    if (t + 1 < L_) {
      for (int i = tid; i < B_ * E_; i += 256) {
        int b = i >> 8, e = i & 255;
        Abuf[b * A_STRIDE + e] = (__bf16)x_emb[b * (L_ * E_) + (t + 1) * E_ + e];
      }
    }
    __syncthreads();
  }
}

extern "C" void kernel_launch(void* const* d_in, const int* in_sizes, int n_in,
                              void* d_out, int out_size, void* d_ws, size_t ws_size,
                              hipStream_t stream) {
  (void)in_sizes; (void)n_in; (void)out_size; (void)ws_size;
  const float* x_emb = (const float*)d_in[0];
  const float* W_ih  = (const float*)d_in[1];
  const float* W_hh  = (const float*)d_in[2];
  const float* b_ih  = (const float*)d_in[3];
  const float* b_hh  = (const float*)d_in[4];
  const float* Wi    = (const float*)d_in[5];
  const float* bi    = (const float*)d_in[6];
  const float* Wo    = (const float*)d_in[7];
  const float* bo    = (const float*)d_in[8];
  float* out = (float*)d_out;
  char* ws = (char*)d_ws;

  hipMemsetAsync(ws, 0, ZERO_END, stream);
  dnc_prep<<<dim3(128), dim3(256), 0, stream>>>(Wi, Wo, (__bf16*)(ws + OFF_WIBF), (__bf16*)(ws + OFF_WOBF));
  dnc_persistent<<<dim3(NWG), dim3(256), LDS_TOTAL, stream>>>(
      x_emb, W_ih, W_hh, b_ih, b_hh, bi, bo, out, ws);
}
